// CodeDistKLLoss_67697274520236
// MI455X (gfx1250) — compile-verified, matrix operand order checked by reading, not run on hardware
//
#include <hip/hip_runtime.h>
#include <hip/hip_bf16.h>
#include <math.h>

#define NC 4096
#define TD 512
#define AZ_STRIDE 536   // 1072B row stride: 16B-aligned, bank-conflict-free across 16 rows
#define M_TILE 48       // 3 WMMA M-subtiles per block: 3x B reuse per load
#define EPS_ 1e-8f

typedef _Float16 v16h __attribute__((ext_vector_type(16)));
typedef _Float16 v8h  __attribute__((ext_vector_type(8)));
typedef float    v8f  __attribute__((ext_vector_type(8)));

// ---------------------------------------------------------------------------
// Kernel 0: zero the accumulators (ws is poisoned by harness, not zeroed)
// ---------------------------------------------------------------------------
__global__ void init_kernel(float* sdist, float* tcnt, float* csum) {
    int i = blockIdx.x * blockDim.x + threadIdx.x;
    if (i < NC) { sdist[i] = 0.0f; tcnt[i] = 0.0f; }
    if (i == 0) csum[0] = 0.0f;
}

// ---------------------------------------------------------------------------
// Kernel 1: codebook f32 -> f16, per-code squared norm, sum of norms
// ---------------------------------------------------------------------------
__global__ void prep_codebook(const float* __restrict__ cb, _Float16* __restrict__ ch,
                              float* __restrict__ csq, float* __restrict__ csum) {
    __shared__ float red[128];
    const int c = blockIdx.x;
    const int tid = threadIdx.x;
    float s = 0.0f;
    for (int k = tid; k < TD; k += 128) {
        float v = cb[(size_t)c * TD + k];
        ch[(size_t)c * TD + k] = (_Float16)v;
        s += v * v;
    }
    red[tid] = s; __syncthreads();
    for (int off = 64; off > 0; off >>= 1) {
        if (tid < off) red[tid] += red[tid + off];
        __syncthreads();
    }
    if (tid == 0) { csq[c] = red[0]; atomicAdd(csum, red[0]); }
}

// ---------------------------------------------------------------------------
// Kernel 2: transpose [B, D, T] f32 -> zh [B*T, D] f16 (32x32 LDS tiles)
// ---------------------------------------------------------------------------
__global__ void prep_features(const float* __restrict__ sf, _Float16* __restrict__ zh,
                              int D, int T) {
    __shared__ float tile[32][33];
    const int tTiles = (T + 31) / 32;
    const int dTiles = D / 32;
    int bid = blockIdx.x;
    const int b   = bid / (dTiles * tTiles);
    int rem       = bid % (dTiles * tTiles);
    const int d0  = (rem / tTiles) * 32;
    const int t0  = (rem % tTiles) * 32;
    const int tid = threadIdx.x;
    const int lx = tid & 31, ly = tid >> 5;      // 32 x 8 threads
#pragma unroll
    for (int i = 0; i < 4; i++) {
        int d = d0 + ly + i * 8;
        int t = t0 + lx;
        tile[ly + i * 8][lx] = (t < T) ? sf[((size_t)b * D + d) * T + t] : 0.0f;
    }
    __syncthreads();
#pragma unroll
    for (int i = 0; i < 4; i++) {
        int t = t0 + ly + i * 8;
        int d = d0 + lx;
        if (t < T) zh[((size_t)b * T + t) * TD + d] = (_Float16)tile[lx][ly + i * 8];
    }
}

// ---------------------------------------------------------------------------
// Kernel 3: per-frame squared norm (from f16 copy) + length mask
// ---------------------------------------------------------------------------
__global__ void row_stats(const _Float16* __restrict__ zh, const int* __restrict__ lengths,
                          const int* __restrict__ stridePtr,
                          float* __restrict__ zsq, float* __restrict__ maskv, int T) {
    __shared__ float red[128];
    const int n = blockIdx.x;
    const int tid = threadIdx.x;
    float s = 0.0f;
    for (int k = tid; k < TD; k += 128) {
        float v = (float)zh[(size_t)n * TD + k];
        s += v * v;
    }
    red[tid] = s; __syncthreads();
    for (int off = 64; off > 0; off >>= 1) {
        if (tid < off) red[tid] += red[tid + off];
        __syncthreads();
    }
    if (tid == 0) {
        zsq[n] = red[0];
        int t = n % T, b = n / T;
        maskv[n] = (t * stridePtr[0] < lengths[b]) ? 1.0f : 0.0f;
    }
}

// ---------------------------------------------------------------------------
// Kernel 4: teacher masked histogram
// ---------------------------------------------------------------------------
__global__ void teacher_hist(const int* __restrict__ codes, const int* __restrict__ lengths,
                             const int* __restrict__ stridePtr, float* __restrict__ tcnt,
                             int N, int T) {
    int n = blockIdx.x * blockDim.x + threadIdx.x;
    if (n >= N) return;
    int t = n % T, b = n / T;
    if (t * stridePtr[0] < lengths[b]) {
        int c = codes[n];
        if (c >= 0 && c < NC) atomicAdd(&tcnt[c], 1.0f);
    }
}

// ---------------------------------------------------------------------------
// Kernel 5: fused cdist + softmax + masked column-sum.
// One block = 48 frames x all 4096 codes; each wave owns a 16-code tile,
// loads each B fragment once, and issues 3 WMMAs (3 M-subtiles from LDS).
// Phase 1: softmax denominators. Phase 2: recompute, accumulate mask*prob.
// ---------------------------------------------------------------------------
__global__ __launch_bounds__(256) void gemm_softmax(
    const _Float16* __restrict__ zh, const float* __restrict__ zsq,
    const float* __restrict__ maskv, const _Float16* __restrict__ ch,
    const float* __restrict__ csq, const float* __restrict__ csum,
    float* __restrict__ sdist, int Nrows) {

    __shared__ __align__(16) _Float16 Az[M_TILE][AZ_STRIDE];
    __shared__ float sZsq[M_TILE], sShift[M_TILE], sDen[M_TILE], sScale[M_TILE];

    const int tid = threadIdx.x;
    const int m0 = blockIdx.x * M_TILE;

    // Stage A-tile (48 rows x 512 f16) into LDS: 64B per thread per row-pass.
    {
        int off = (tid & 15) * 32;   // halves
#pragma unroll
        for (int rp = 0; rp < M_TILE / 16; rp++) {
            int row = (tid >> 4) + rp * 16;
            int gr = m0 + row; if (gr > Nrows - 1) gr = Nrows - 1;
            const uint4* src = (const uint4*)(zh + (size_t)gr * TD + off);
            uint4* dst = (uint4*)(&Az[row][off]);
            dst[0] = src[0]; dst[1] = src[1]; dst[2] = src[2]; dst[3] = src[3];
        }
    }
    if (tid < M_TILE) {
        int gr = m0 + tid; if (gr > Nrows - 1) gr = Nrows - 1;
        float zq = zsq[gr];
        sZsq[tid]   = zq;
        sShift[tid] = sqrtf(fmaxf(zq + csum[0] * (1.0f / NC), 0.0f));
        sDen[tid]   = 0.0f;
    }
    __syncthreads();

    const int wave = tid >> 5;
    const int lane = tid & 31;
    const int halfSel = lane >> 4;
    const int ln = lane & 15;
    const _Float16* arow0 = &Az[ln][0];
    const _Float16* arow1 = &Az[16 + ln][0];
    const _Float16* arow2 = &Az[32 + ln][0];

    // ---- Phase 1: denominators ----
    float denLoc[24];
#pragma unroll
    for (int r = 0; r < 24; r++) denLoc[r] = 0.0f;

    for (int nt = wave; nt < NC / 16; nt += 8) {
        const int n0 = nt * 16;
        const _Float16* brow = ch + (size_t)(n0 + ln) * TD;
        v8f acc0 = {}, acc1 = {}, acc2 = {};
#pragma unroll
        for (int k0 = 0; k0 < TD; k0 += 32) {
            const int ka = k0 + halfSel * 8;
            v8h blo = *(const v8h*)(brow + ka);
            v8h bhi = *(const v8h*)(brow + ka + 16);
            v16h b = __builtin_shufflevector(blo, bhi, 0,1,2,3,4,5,6,7,8,9,10,11,12,13,14,15);
            v8h a0l = *(const v8h*)(arow0 + ka);
            v8h a0h = *(const v8h*)(arow0 + ka + 16);
            v16h a0 = __builtin_shufflevector(a0l, a0h, 0,1,2,3,4,5,6,7,8,9,10,11,12,13,14,15);
            acc0 = __builtin_amdgcn_wmma_f32_16x16x32_f16(false, a0, false, b, (short)0, acc0, false, false);
            v8h a1l = *(const v8h*)(arow1 + ka);
            v8h a1h = *(const v8h*)(arow1 + ka + 16);
            v16h a1 = __builtin_shufflevector(a1l, a1h, 0,1,2,3,4,5,6,7,8,9,10,11,12,13,14,15);
            acc1 = __builtin_amdgcn_wmma_f32_16x16x32_f16(false, a1, false, b, (short)0, acc1, false, false);
            v8h a2l = *(const v8h*)(arow2 + ka);
            v8h a2h = *(const v8h*)(arow2 + ka + 16);
            v16h a2 = __builtin_shufflevector(a2l, a2h, 0,1,2,3,4,5,6,7,8,9,10,11,12,13,14,15);
            acc2 = __builtin_amdgcn_wmma_f32_16x16x32_f16(false, a2, false, b, (short)0, acc2, false, false);
        }
        float cn = csq[n0 + ln];
#pragma unroll
        for (int r = 0; r < 8; r++) {
            int mr0 = halfSel * 8 + r;
            float sq0 = sZsq[mr0] + cn - 2.0f * acc0[r];
            denLoc[r]      += __expf(sShift[mr0] - sqrtf(fmaxf(sq0, 0.0f)));
            int mr1 = 16 + mr0;
            float sq1 = sZsq[mr1] + cn - 2.0f * acc1[r];
            denLoc[8 + r]  += __expf(sShift[mr1] - sqrtf(fmaxf(sq1, 0.0f)));
            int mr2 = 32 + mr0;
            float sq2 = sZsq[mr2] + cn - 2.0f * acc2[r];
            denLoc[16 + r] += __expf(sShift[mr2] - sqrtf(fmaxf(sq2, 0.0f)));
        }
    }
#pragma unroll
    for (int m = 0; m < 3; m++)
#pragma unroll
        for (int r = 0; r < 8; r++)
            atomicAdd(&sDen[m * 16 + halfSel * 8 + r], denLoc[m * 8 + r]);
    __syncthreads();
    if (tid < M_TILE) {
        float sc = 0.0f;
        if (m0 + tid < Nrows) sc = maskv[m0 + tid] / sDen[tid];
        sScale[tid] = sc;
    }
    __syncthreads();

    // ---- Phase 2: masked probabilities -> per-column accumulation ----
    for (int nt = wave; nt < NC / 16; nt += 8) {
        const int n0 = nt * 16;
        const _Float16* brow = ch + (size_t)(n0 + ln) * TD;
        v8f acc0 = {}, acc1 = {}, acc2 = {};
#pragma unroll
        for (int k0 = 0; k0 < TD; k0 += 32) {
            const int ka = k0 + halfSel * 8;
            v8h blo = *(const v8h*)(brow + ka);
            v8h bhi = *(const v8h*)(brow + ka + 16);
            v16h b = __builtin_shufflevector(blo, bhi, 0,1,2,3,4,5,6,7,8,9,10,11,12,13,14,15);
            v8h a0l = *(const v8h*)(arow0 + ka);
            v8h a0h = *(const v8h*)(arow0 + ka + 16);
            v16h a0 = __builtin_shufflevector(a0l, a0h, 0,1,2,3,4,5,6,7,8,9,10,11,12,13,14,15);
            acc0 = __builtin_amdgcn_wmma_f32_16x16x32_f16(false, a0, false, b, (short)0, acc0, false, false);
            v8h a1l = *(const v8h*)(arow1 + ka);
            v8h a1h = *(const v8h*)(arow1 + ka + 16);
            v16h a1 = __builtin_shufflevector(a1l, a1h, 0,1,2,3,4,5,6,7,8,9,10,11,12,13,14,15);
            acc1 = __builtin_amdgcn_wmma_f32_16x16x32_f16(false, a1, false, b, (short)0, acc1, false, false);
            v8h a2l = *(const v8h*)(arow2 + ka);
            v8h a2h = *(const v8h*)(arow2 + ka + 16);
            v16h a2 = __builtin_shufflevector(a2l, a2h, 0,1,2,3,4,5,6,7,8,9,10,11,12,13,14,15);
            acc2 = __builtin_amdgcn_wmma_f32_16x16x32_f16(false, a2, false, b, (short)0, acc2, false, false);
        }
        float cn = csq[n0 + ln];
        float colAcc = 0.0f;
#pragma unroll
        for (int r = 0; r < 8; r++) {
            int mr0 = halfSel * 8 + r;
            float sq0 = sZsq[mr0] + cn - 2.0f * acc0[r];
            colAcc += __expf(sShift[mr0] - sqrtf(fmaxf(sq0, 0.0f))) * sScale[mr0];
            int mr1 = 16 + mr0;
            float sq1 = sZsq[mr1] + cn - 2.0f * acc1[r];
            colAcc += __expf(sShift[mr1] - sqrtf(fmaxf(sq1, 0.0f))) * sScale[mr1];
            int mr2 = 32 + mr0;
            float sq2 = sZsq[mr2] + cn - 2.0f * acc2[r];
            colAcc += __expf(sShift[mr2] - sqrtf(fmaxf(sq2, 0.0f))) * sScale[mr2];
        }
        colAcc += __shfl_xor(colAcc, 16, 32);
        if (halfSel == 0) atomicAdd(&sdist[n0 + ln], colAcc);
    }
}

// ---------------------------------------------------------------------------
// Kernel 6: normalize both distributions and reduce the KL scalar
// ---------------------------------------------------------------------------
__global__ void finalize_kl(const float* __restrict__ sdist, const float* __restrict__ tcnt,
                            float* __restrict__ out) {
    __shared__ float redA[256], redB[256];
    __shared__ float S, Tt;
    const int tid = threadIdx.x;
    float sa = 0.0f, sb = 0.0f;
    for (int c = tid; c < NC; c += 256) { sa += sdist[c]; sb += tcnt[c]; }
    redA[tid] = sa; redB[tid] = sb; __syncthreads();
    for (int off = 128; off > 0; off >>= 1) {
        if (tid < off) { redA[tid] += redA[tid + off]; redB[tid] += redB[tid + off]; }
        __syncthreads();
    }
    if (tid == 0) { S = redA[0] + EPS_; Tt = redB[0] + EPS_; }
    __syncthreads();
    float kl = 0.0f;
    for (int c = tid; c < NC; c += 256) {
        float s = sdist[c] / S;
        float t = tcnt[c] / Tt;
        kl += s * logf(s + EPS_) - s * logf(t + EPS_);
    }
    redA[tid] = kl; __syncthreads();
    for (int off = 128; off > 0; off >>= 1) {
        if (tid < off) redA[tid] += redA[tid + off];
        __syncthreads();
    }
    if (tid == 0) out[0] = redA[0];
}

// ---------------------------------------------------------------------------
// Host launcher
// ---------------------------------------------------------------------------
static inline size_t align256(size_t x) { return (x + 255) & ~(size_t)255; }

extern "C" void kernel_launch(void* const* d_in, const int* in_sizes, int n_in,
                              void* d_out, int out_size, void* d_ws, size_t ws_size,
                              hipStream_t stream) {
    const float* sf      = (const float*)d_in[0];   // [B, D, T]
    const int*   codes   = (const int*)d_in[1];     // [B, T]
    const float* cb      = (const float*)d_in[2];   // [NC, D]
    const int*   lengths = (const int*)d_in[3];     // [B]
    const int*   stride  = (const int*)d_in[4];     // scalar (device)

    const int N = in_sizes[1];            // B*T = 12000
    const int B = in_sizes[3];            // 8
    const int T = N / B;                  // 1500
    const int D = in_sizes[0] / N;        // 512 (== TD)

    // Workspace carve-up
    char* ws = (char*)d_ws;
    size_t off = 0;
    _Float16* zh   = (_Float16*)(ws + off); off = align256(off + (size_t)N * TD * 2);
    float*    zsq  = (float*)(ws + off);    off = align256(off + (size_t)N * 4);
    float*    mskv = (float*)(ws + off);    off = align256(off + (size_t)N * 4);
    _Float16* ch   = (_Float16*)(ws + off); off = align256(off + (size_t)NC * TD * 2);
    float*    csq  = (float*)(ws + off);    off = align256(off + (size_t)NC * 4);
    float*    csum = (float*)(ws + off);    off = align256(off + 16);
    float*    sdst = (float*)(ws + off);    off = align256(off + (size_t)NC * 4);
    float*    tcnt = (float*)(ws + off);    off = align256(off + (size_t)NC * 4);
    (void)ws_size; (void)n_in; (void)out_size; (void)D;

    init_kernel<<<(NC + 255) / 256, 256, 0, stream>>>(sdst, tcnt, csum);
    prep_codebook<<<NC, 128, 0, stream>>>(cb, ch, csq, csum);
    {
        int tTiles = (T + 31) / 32;
        int dTiles = TD / 32;
        prep_features<<<B * dTiles * tTiles, 256, 0, stream>>>(sf, zh, TD, T);
    }
    row_stats<<<N, 128, 0, stream>>>(zh, lengths, stride, zsq, mskv, T);
    teacher_hist<<<(N + 255) / 256, 256, 0, stream>>>(codes, lengths, stride, tcnt, N, T);
    gemm_softmax<<<(N + M_TILE - 1) / M_TILE, 256, 0, stream>>>(zh, zsq, mskv, ch, csq, csum,
                                                                sdst, N);
    finalize_kl<<<1, 256, 0, stream>>>(sdst, tcnt, (float*)d_out);
}
